// CorrelationNetwork_64269890618022
// MI455X (gfx1250) — compile-verified, matrix operand order checked by reading, not run on hardware
//
#include <hip/hip_runtime.h>
#include <hip/hip_bf16.h>

typedef __attribute__((ext_vector_type(16))) _Float16 v16h;
typedef __attribute__((ext_vector_type(8)))  float    v8f;

#define BATCH 128
#define NAG   64      // agents
#define HDIM  128     // hidden

// ---------------------------------------------------------------------------
// Kernel 1: Ai[b,n,h] = sum_f feat[b,n,f]*W1[f,h];  Aj uses W1[64+f,h]
// grid = B*N blocks, 128 threads (one per h)
// ---------------------------------------------------------------------------
__global__ __launch_bounds__(128) void proj_kernel(
    const float* __restrict__ feat, const float* __restrict__ W1,
    float* __restrict__ Ai, float* __restrict__ Aj)
{
    __shared__ float sf[NAG];
    const int bn = blockIdx.x;           // 0 .. B*N-1
    const int h  = threadIdx.x;          // 0 .. 127
    const float* fr = feat + (size_t)bn * 128;   // F = 128, use first 64
    if (h < NAG) sf[h] = fr[h];
    __syncthreads();
    float a1 = 0.f, a2 = 0.f;
    #pragma unroll 8
    for (int f = 0; f < NAG; ++f) {
        const float fv = sf[f];
        a1 = fmaf(fv, W1[f * HDIM + h], a1);
        a2 = fmaf(fv, W1[(NAG + f) * HDIM + h], a2);
    }
    Ai[(size_t)bn * HDIM + h] = a1;
    Aj[(size_t)bn * HDIM + h] = a2;
}

// ---------------------------------------------------------------------------
// helper: K index inside a 32-wide block for element e of a v16h fragment,
// per CDNA5 16-bit A/B VGPR layout (VGPR0..3 -> K 0..7 / VGPR4..7 -> K 16..23,
// the other K-half selected by lane>=16).
// ---------------------------------------------------------------------------
__device__ __forceinline__ int frag_k(int e) {
    const int v = e >> 1, p = e & 1;
    return (v < 4) ? (2 * v + p) : (16 + 2 * (v - 4) + p);
}

// ---------------------------------------------------------------------------
// Kernel 2: fused h1 -> (h1@W2) -> relu -> dot(w3) -> sigmoid
// block = 256 threads (8 waves). Wave w handles i = i_base+w, j = j_base+0..15.
// blockIdx.x = b*32 + g ; g: ib = g>>2 (8 i-groups of 8), jb = g&3 (4 j-groups of 16)
// ---------------------------------------------------------------------------
__global__ __launch_bounds__(256) void corr_main(
    const float* __restrict__ Ai, const float* __restrict__ Aj,
    const float* __restrict__ b1, const float* __restrict__ W2,
    const float* __restrict__ b2, const float* __restrict__ w3,
    const float* __restrict__ b3, float* __restrict__ out)
{
    __shared__ float    sAi[8  * HDIM];            // 4 KB
    __shared__ float    sAj[16 * HDIM];            // 8 KB
    __shared__ _Float16 sW2s[8 * 4 * 32 * 16];     // 32 KB, fragment-swizzled
    __shared__ float    sB1[HDIM], sB2[HDIM], sW3[HDIM];

    const int tid = threadIdx.x;
    const int g   = blockIdx.x & 31;
    const int b   = blockIdx.x >> 5;
    const int i_base = (g >> 2) * 8;
    const int j_base = (g & 3) * 16;

    // ---- stage W2 into LDS, pre-swizzled into WMMA B-fragment order -------
    // dest index d = (((t*4 + kb)*32 + lane)*16 + e)
    for (int d = tid; d < 8 * 4 * 32 * 16; d += 256) {
        const int e    = d & 15;
        const int lane = (d >> 4) & 31;
        const int kb   = (d >> 9) & 3;
        const int t    = d >> 11;
        const int n    = t * 16 + (lane & 15);
        const int k    = kb * 32 + (lane >> 4) * 8 + frag_k(e);
        sW2s[d] = (_Float16)W2[k * HDIM + n];      // W2 is [h(k), out(n)]
    }
    // ---- stage Ai (8 rows) and Aj (16 rows) --------------------------------
    const float* AiB = Ai + ((size_t)b * NAG + i_base) * HDIM;
    const float* AjB = Aj + ((size_t)b * NAG + j_base) * HDIM;
    for (int d = tid; d < 8 * HDIM;  d += 256) sAi[d] = AiB[d];
    for (int d = tid; d < 16 * HDIM; d += 256) sAj[d] = AjB[d];
    if (tid < HDIM) { sB1[tid] = b1[tid]; sB2[tid] = b2[tid]; sW3[tid] = w3[tid]; }
    __syncthreads();

    const int wave  = tid >> 5;          // 0..7  -> i = i_base + wave
    const int lane  = tid & 31;
    const int lm    = lane & 15;         // A-row / B-col / C-col index
    const int khalf = (lane >> 4) * 8;   // K-half select

    // ---- build A fragments: h1 row = relu(Ai[i] + Aj[j] + b1) in f16 -------
    const float* aiR = &sAi[wave * HDIM];
    const float* ajR = &sAj[lm   * HDIM];
    v16h afrag[4];
    #pragma unroll
    for (int kb = 0; kb < 4; ++kb) {
        const int k0 = kb * 32 + khalf;
        #pragma unroll
        for (int e = 0; e < 16; ++e) {
            const int k = k0 + frag_k(e);
            float x = aiR[k] + ajR[k] + sB1[k];
            afrag[kb][e] = (_Float16)(x > 0.f ? x : 0.f);
        }
    }

    // ---- 8 N-tiles: 4 chained WMMAs each, then relu(+b2) and dot(w3) -------
    float dot[8] = {0.f, 0.f, 0.f, 0.f, 0.f, 0.f, 0.f, 0.f};
    for (int t = 0; t < 8; ++t) {
        v8f c = {};
        #pragma unroll
        for (int kb = 0; kb < 4; ++kb) {
            const v16h bfrag =
                *(const v16h*)&sW2s[(((t * 4 + kb) * 32) + lane) * 16];
            c = __builtin_amdgcn_wmma_f32_16x16x32_f16(
                    false, afrag[kb], false, bfrag, (short)0, c, false, false);
        }
        const int   n  = t * 16 + lm;
        const float bb = sB2[n];
        const float ww = sW3[n];
        #pragma unroll
        for (int rv = 0; rv < 8; ++rv) {
            float h2v = c[rv] + bb;
            h2v = h2v > 0.f ? h2v : 0.f;
            dot[rv] = fmaf(h2v, ww, dot[rv]);
        }
    }

    // ---- reduce over the 16 columns held per half-wave ---------------------
    #pragma unroll
    for (int rv = 0; rv < 8; ++rv) {
        float s = dot[rv];
        s += __shfl_xor(s, 1, 32);
        s += __shfl_xor(s, 2, 32);
        s += __shfl_xor(s, 4, 32);
        s += __shfl_xor(s, 8, 32);
        dot[rv] = s;
    }

    // lane 0 holds rows (j) j_base+0..7 ; lane 16 holds j_base+8..15
    if (lm == 0) {
        const float b3v = b3[0];
        const int i = i_base + wave;
        float* o = out + ((size_t)b * NAG + i) * NAG + j_base + khalf;
        #pragma unroll
        for (int rv = 0; rv < 8; ++rv) {
            const float x = dot[rv] + b3v;
            o[rv] = 1.f / (1.f + __expf(-x));
        }
    }
}

// ---------------------------------------------------------------------------
// Kernel 3: softmax over 64 mixing weights
// ---------------------------------------------------------------------------
__global__ __launch_bounds__(64) void softmax64(
    const float* __restrict__ mw, float* __restrict__ out)
{
    const int t = threadIdx.x;           // 64 threads
    float mx = -3.402823466e+38f;
    for (int k = 0; k < NAG; ++k) mx = fmaxf(mx, mw[k]);
    float s = 0.f;
    for (int k = 0; k < NAG; ++k) s += __expf(mw[k] - mx);
    out[t] = __expf(mw[t] - mx) / s;
}

extern "C" void kernel_launch(void* const* d_in, const int* in_sizes, int n_in,
                              void* d_out, int out_size, void* d_ws, size_t ws_size,
                              hipStream_t stream) {
    (void)in_sizes; (void)n_in; (void)out_size; (void)ws_size;
    const float* feat = (const float*)d_in[0];   // (128,64,128)
    const float* W1   = (const float*)d_in[1];   // (128,128)
    const float* b1   = (const float*)d_in[2];   // (128,)
    const float* W2   = (const float*)d_in[3];   // (128,128)
    const float* b2   = (const float*)d_in[4];   // (128,)
    const float* w3   = (const float*)d_in[5];   // (128,)
    const float* b3   = (const float*)d_in[6];   // (1,)
    const float* mw   = (const float*)d_in[7];   // (64,)
    float* out = (float*)d_out;

    float* Ai = (float*)d_ws;                       // 128*64*128 f32 = 4 MB
    float* Aj = Ai + (size_t)BATCH * NAG * HDIM;    // +4 MB

    proj_kernel<<<BATCH * NAG, 128, 0, stream>>>(feat, W1, Ai, Aj);
    corr_main<<<BATCH * 32, 256, 0, stream>>>(Ai, Aj, b1, W2, b2, w3, b3, out);
    softmax64<<<1, 64, 0, stream>>>(mw, out + (size_t)BATCH * NAG * NAG);
}
